// TransformerModule_28802050687839
// MI455X (gfx1250) — compile-verified, matrix operand order checked by reading, not run on hardware
//
#include <hip/hip_runtime.h>
#include <hip/hip_bf16.h>

// ---------------------------------------------------------------------------
// Transformer-XL forward for MI455X (gfx1250, wave32, WMMA).
// All GEMMs: f16 operands, f32 accumulate via v_wmma_f32_16x16x32_f16,
// 2x2 register-blocked per wave (32x32 out/wave, 2 b128 loads per WMMA).
// ---------------------------------------------------------------------------

typedef __attribute__((ext_vector_type(16))) _Float16 v16h;
typedef __attribute__((ext_vector_type(8)))  _Float16 v8h;
typedef __attribute__((ext_vector_type(8)))  float    v8f;

// Model constants
#define QLEN   512
#define MLEN   128
#define KLEN   640          // MLEN + QLEN
#define BSZ    4
#define NH     16
#define DH     64
#define DM     1024
#define DFF    4096
#define NL     6
#define TOKQ   (QLEN*BSZ)   // 2048
#define TOKK   (KLEN*BSZ)   // 2560

// Load one 16xK=32 f16 fragment chunk pair per the gfx1250 16-bit A/B VGPR
// layout: lanes 0-15 hold K {0..7, 16..23}, lanes 16-31 hold K {8..15, 24..31}
// (caller bakes the +8 half-offset into the pointer).
static __device__ __forceinline__ v16h load_frag(const _Float16* __restrict__ p)
{
    v8h lo = *(const v8h*)(p);
    v8h hi = *(const v8h*)(p + 16);
    v16h r;
#pragma unroll
    for (int e = 0; e < 8; ++e) { r[e] = lo[e]; r[e+8] = hi[e]; }
    return r;
}

static __device__ __forceinline__ v8f wmma_f16(v16h a, v16h b, v8f c)
{
    return __builtin_amdgcn_wmma_f32_16x16x32_f16(
               false, a, false, b, (short)0, c, false, false);
}

// ---------------------------------------------------------------------------
// Generic strided/batched WMMA GEMM:  C[M,N] = epi( A[M,K] * B[N,K]^T )
// A, B f16 row-major (leading dims lda/ldb in elements).
// Block = 256 threads = 8 waves as 2(M) x 4(N); each wave owns a 32x32 tile
// (2x2 of 16x16 WMMA tiles) -> block tile 64(M) x 128(N).
// Requires: M % 64 == 0, N % 32 == 0 (waves with tileN >= N whole-wave exit).
// z-batch offsets: off = Div*(z/zdiv) + Mod*(z%zdiv)   (elements).
//
// EPI: 0 = store f16
//      1 = store f32
//      2 = relu(c + bias[n]) -> f16
//      3 = c + (bias?bias[n]:0) + resid[ci] -> f32 (fused residual add)
//      4 = Transformer-XL rel-shift scatter-add:
//          j = n - (M-1-m); if j>=0: C[m, j] += c   (C is f32)
// ---------------------------------------------------------------------------
template<int EPI>
__global__ __launch_bounds__(256) void gemm_wmma(
    const _Float16* __restrict__ A, const _Float16* __restrict__ B,
    void* __restrict__ Cv,
    const float* __restrict__ bias, const float* __restrict__ resid,
    int M, int N, int K, int lda, int ldb, int ldc, int zdiv,
    long long aDiv, long long aMod, long long bDiv, long long bMod,
    long long cDiv, long long cMod)
{
    const int z  = blockIdx.z;
    const int zq = z / zdiv, zr = z % zdiv;
    const _Float16* Ab = A + aDiv*(long long)zq + aMod*(long long)zr;
    const _Float16* Bb = B + bDiv*(long long)zq + bMod*(long long)zr;
    const long long cOff = cDiv*(long long)zq + cMod*(long long)zr;

    const int wave  = threadIdx.x >> 5;
    const int lane  = threadIdx.x & 31;
    const int tileM = blockIdx.y*64  + (wave >> 2)*32;
    const int tileN = blockIdx.x*128 + (wave &  3)*32;
    if (tileN >= N) return;                    // whole-wave exit (no barriers)
    const int r15   = lane & 15;
    const int hofs  = (lane & 16) ? 8 : 0;

    const _Float16* a0 = Ab + (long long)(tileM + r15)*lda + hofs;
    const _Float16* a1 = a0 + (long long)16*lda;
    const _Float16* b0 = Bb + (long long)(tileN + r15)*ldb + hofs;
    const _Float16* b1 = b0 + (long long)16*ldb;

    v8f acc[2][2] = {};
    for (int k0 = 0; k0 < K; k0 += 32) {
        v16h va0 = load_frag(a0 + k0);
        v16h va1 = load_frag(a1 + k0);
        v16h vb0 = load_frag(b0 + k0);
        v16h vb1 = load_frag(b1 + k0);
        acc[0][0] = wmma_f16(va0, vb0, acc[0][0]);
        acc[0][1] = wmma_f16(va0, vb1, acc[0][1]);
        acc[1][0] = wmma_f16(va1, vb0, acc[1][0]);
        acc[1][1] = wmma_f16(va1, vb1, acc[1][1]);
    }

    // C layout per tile: VGPR r -> M = base + r + (lane>=16 ? 8 : 0),
    //                    N = base + lane&15
#pragma unroll
    for (int mt = 0; mt < 2; ++mt) {
#pragma unroll
        for (int nt = 0; nt < 2; ++nt) {
            const int nIdx = tileN + nt*16 + r15;
#pragma unroll
            for (int r = 0; r < 8; ++r) {
                const int m = tileM + mt*16 + hofs + r;
                const float c = acc[mt][nt][r];
                const long long ci = cOff + (long long)m*ldc + nIdx;
                if constexpr (EPI == 0) {
                    ((_Float16*)Cv)[ci] = (_Float16)c;
                } else if constexpr (EPI == 1) {
                    ((float*)Cv)[ci] = c;
                } else if constexpr (EPI == 2) {
                    float t = c + bias[nIdx];
                    ((_Float16*)Cv)[ci] = (_Float16)(t > 0.f ? t : 0.f);
                } else if constexpr (EPI == 3) {
                    float t = c + resid[ci];
                    if (bias) t += bias[nIdx];
                    ((float*)Cv)[ci] = t;
                } else if constexpr (EPI == 4) {
                    const int j = nIdx - (M - 1 - m);  // rel_shift
                    if (j >= 0)
                        ((float*)Cv)[cOff + (long long)m*ldc + j] += c;
                }
            }
        }
    }
}

// ---------------------------------------------------------------------------
// LayerNorm over D=1024, one 128-thread block per row.
// If memsl != null: concat-aware ([mems_l ; x] along klen).
// Writes f16 (out16) or f32 (out32).
// ---------------------------------------------------------------------------
__global__ __launch_bounds__(128) void ln_kernel(
    const float* __restrict__ memsl, const float* __restrict__ x,
    const float* __restrict__ g, const float* __restrict__ b,
    _Float16* __restrict__ out16, float* __restrict__ out32,
    int mlen, int bsz)
{
    const int row = blockIdx.x;
    const int kk = row / bsz, bb = row % bsz;
    const float* src;
    if (memsl != nullptr && kk < mlen) {
        src = memsl + (long long)row * DM;                  // (kk*bsz+bb)*DM
    } else {
        const int k2 = (memsl != nullptr) ? (kk - mlen) : kk;
        src = x + ((long long)k2 * bsz + bb) * DM;
    }
    float v[8]; float s = 0.f, s2 = 0.f;
#pragma unroll
    for (int c = 0; c < 8; ++c) {
        float t = src[threadIdx.x + 128*c];
        v[c] = t; s += t; s2 += t*t;
    }
    __shared__ float r1[128], r2[128];
    r1[threadIdx.x] = s; r2[threadIdx.x] = s2; __syncthreads();
    for (int st = 64; st > 0; st >>= 1) {
        if (threadIdx.x < st) {
            r1[threadIdx.x] += r1[threadIdx.x + st];
            r2[threadIdx.x] += r2[threadIdx.x + st];
        }
        __syncthreads();
    }
    const float mu  = r1[0] * (1.f/DM);
    const float var = r2[0] * (1.f/DM) - mu*mu;
    const float ri  = rsqrtf(var + 1e-5f);
#pragma unroll
    for (int c = 0; c < 8; ++c) {
        const int d = threadIdx.x + 128*c;
        const float y = (v[c] - mu) * ri * g[d] + b[d];
        if (out16) out16[(long long)row*DM + d] = (_Float16)y;
        else       out32[(long long)row*DM + d] = y;
    }
}

// ---------------------------------------------------------------------------
// Masked softmax over klen; reads f32 score row, writes f16 prob IN PLACE
// over the start of the same row (all reads complete before the reduction
// barriers, so the aliasing is safe). Masked: j > i + mlen.
// blockIdx.x = z*QLEN + i, z over BSZ*NH.
// ---------------------------------------------------------------------------
__global__ __launch_bounds__(128) void softmax_kernel(
    float* __restrict__ score, float scale)
{
    const long long row = blockIdx.x;
    const int i = (int)(row % QLEN);
    float* srow = score + row * KLEN;
    _Float16* prow = (_Float16*)srow;

    const int jmax = i + MLEN;               // last unmasked column
    float vals[5];                           // KLEN / 128 = 5
    float m = -INFINITY;
#pragma unroll
    for (int c = 0; c < 5; ++c) {
        const int j = threadIdx.x + 128*c;
        float s = (j <= jmax) ? srow[j] * scale : -INFINITY;
        vals[c] = s;
        m = fmaxf(m, s);
    }
    __shared__ float red[128];
    red[threadIdx.x] = m; __syncthreads();
    for (int st = 64; st > 0; st >>= 1) {
        if (threadIdx.x < st) red[threadIdx.x] = fmaxf(red[threadIdx.x], red[threadIdx.x + st]);
        __syncthreads();
    }
    m = red[0]; __syncthreads();
    float sum = 0.f;
#pragma unroll
    for (int c = 0; c < 5; ++c) {
        float e = __expf(vals[c] - m);
        vals[c] = e; sum += e;
    }
    red[threadIdx.x] = sum; __syncthreads();
    for (int st = 64; st > 0; st >>= 1) {
        if (threadIdx.x < st) red[threadIdx.x] += red[threadIdx.x + st];
        __syncthreads();
    }
    const float inv = 1.f / red[0];
    __syncthreads();
#pragma unroll
    for (int c = 0; c < 5; ++c) {
        const int j = threadIdx.x + 128*c;
        prow[j] = (_Float16)(vals[c] * inv);   // masked -> exp=0 -> prob 0
    }
}

// ---------------------------------------------------------------------------
// Split fused QKV result: add r_w/r_r biases into two f16 Q copies (last
// qlen rows only), copy K to (b,h,klen,dh), transpose V to (b,h,dh,klen).
// One thread per (token, channel): klen*BSZ*DM threads.
// ---------------------------------------------------------------------------
__global__ __launch_bounds__(256) void qkv_prep(
    const _Float16* __restrict__ wh,
    const float* __restrict__ rwb, const float* __restrict__ rrb,
    _Float16* __restrict__ qw, _Float16* __restrict__ qr,
    _Float16* __restrict__ kb, _Float16* __restrict__ vT)
{
    const long long idx = (long long)blockIdx.x*256 + threadIdx.x;
    const int t = (int)(idx >> 10);
    const int c = (int)(idx & 1023);
    const int kk = t / BSZ, bb = t % BSZ;
    const int h = c >> 6, d = c & 63;
    const _Float16* wrow = wh + (long long)t * (3*DM);
    const long long bh = (long long)(bb*NH + h);

    kb[(bh*KLEN + kk)*DH + d] = wrow[DM   + c];
    vT[(bh*DH   + d)*KLEN + kk] = wrow[2*DM + c];
    if (kk >= MLEN) {
        const int i = kk - MLEN;
        const float qv = (float)wrow[c];
        const long long qi = (bh*QLEN + i)*DH + d;
        qw[qi] = (_Float16)(qv + rwb[c]);     // rwb is (NH,DH) flat = c
        qr[qi] = (_Float16)(qv + rrb[c]);
    }
}

// f32 -> f16 weight conversion
__global__ __launch_bounds__(256) void cvt_f16(
    const float* __restrict__ src, _Float16* __restrict__ dst, long long n)
{
    const long long i = (long long)blockIdx.x*256 + threadIdx.x;
    if (i < n) dst[i] = (_Float16)src[i];
}

// Sinusoidal relative position embedding, positions klen-1..0 clamped at 512.
__global__ __launch_bounds__(256) void pos_kernel(_Float16* __restrict__ pos)
{
    const int idx = blockIdx.x*256 + threadIdx.x;
    if (idx >= KLEN*DM) return;
    const int k = idx >> 10, c = idx & 1023;
    const float p = fminf((float)(KLEN - 1 - k), 512.0f);
    const int i = (c < 512) ? c : c - 512;
    // inv_freq = 10000^(-2i/1024)
    const float invf = __expf(-(float)(2*i) * (9.210340371976184f / 1024.0f));
    const float a = p * invf;
    pos[idx] = (_Float16)((c < 512) ? __sinf(a) : __cosf(a));
}

// ---------------------------------------------------------------------------
// Host orchestration
// ---------------------------------------------------------------------------
extern "C" void kernel_launch(void* const* d_in, const int* in_sizes, int n_in,
                              void* d_out, int out_size, void* d_ws, size_t ws_size,
                              hipStream_t stream) {
    (void)in_sizes; (void)n_in; (void)out_size;
    const float* inputs = (const float*)d_in[0];
    const float* mems   = (const float*)d_in[1];
    const float* rwb    = (const float*)d_in[2];
    const float* rrb    = (const float*)d_in[3];
    const float* qkv_w  = (const float*)d_in[4];
    const float* r_w    = (const float*)d_in[5];
    const float* o_w    = (const float*)d_in[6];
    const float* ln1_g  = (const float*)d_in[7];
    const float* ln1_b  = (const float*)d_in[8];
    const float* ff_w1  = (const float*)d_in[9];
    const float* ff_b1  = (const float*)d_in[10];
    const float* ff_w2  = (const float*)d_in[11];
    const float* ff_b2  = (const float*)d_in[12];
    const float* ln2_g  = (const float*)d_in[13];
    const float* ln2_b  = (const float*)d_in[14];
    const float* fin_g  = (const float*)d_in[15];
    const float* fin_b  = (const float*)d_in[16];

    // Workspace carve-up (256B aligned)
    char* w0 = (char*)d_ws;
    char* wp = w0;
    auto alloc = [&](size_t bytes) -> void* {
        void* p = (void*)wp;
        wp += (bytes + 255) & ~(size_t)255;
        return p;
    };
    float*    X    = (float*)   alloc((size_t)TOKQ*DM*4);        // activations (qlen,B,D)
    _Float16* H16  = (_Float16*)alloc((size_t)TOKK*DM*2);        // LN1 output (klen*B,D)
    _Float16* H2   = (_Float16*)alloc((size_t)TOKQ*DM*2);        // LN2 output
    _Float16* WH   = (_Float16*)alloc((size_t)TOKK*3*DM*2);      // fused QKV result
    _Float16* POS  = (_Float16*)alloc((size_t)KLEN*DM*2);        // pos emb
    _Float16* RK   = (_Float16*)alloc((size_t)KLEN*DM*2);        // r_head_k
    _Float16* QW   = (_Float16*)alloc((size_t)BSZ*NH*QLEN*DH*2); // q + r_w_bias
    _Float16* QR   = (_Float16*)alloc((size_t)BSZ*NH*QLEN*DH*2); // q + r_r_bias
    _Float16* KB   = (_Float16*)alloc((size_t)BSZ*NH*KLEN*DH*2); // keys (b,h,k,d)
    _Float16* VT   = (_Float16*)alloc((size_t)BSZ*NH*DH*KLEN*2); // values^T (b,h,d,k)
    _Float16* AV   = (_Float16*)alloc((size_t)TOKQ*DM*2);        // attn_vec (qlen,B,HD)
    float*    SC   = (float*)   alloc((size_t)BSZ*NH*QLEN*KLEN*4); // scores / prob / FFN hidden
    _Float16* SQKV = (_Float16*)alloc((size_t)3*DM*DM*2);        // staged f16 weights
    _Float16* SRW  = (_Float16*)alloc((size_t)DM*DM*2);
    _Float16* SOW  = (_Float16*)alloc((size_t)DM*DM*2);
    _Float16* SW1  = (_Float16*)alloc((size_t)DFF*DM*2);
    _Float16* SW2  = (_Float16*)alloc((size_t)DM*DFF*2);
    _Float16* FFH  = (_Float16*)SC;                              // FFN hidden aliases SC
    if ((size_t)(wp - w0) > ws_size) return;                     // not enough scratch

    const dim3 blk(256);
    const long long qz = (long long)QLEN*DH;      // per-(b,h) q slab
    const long long kz = (long long)KLEN*DH;      // per-(b,h) k/v slab
    const long long sz = (long long)QLEN*KLEN;    // per-(b,h) score slab

    // x = inputs
    hipMemcpyAsync(X, inputs, (size_t)TOKQ*DM*4, hipMemcpyDeviceToDevice, stream);
    pos_kernel<<<(KLEN*DM + 255)/256, blk, 0, stream>>>(POS);

    for (int l = 0; l < NL; ++l) {
        // Stage this layer's weights as f16
        cvt_f16<<<(3*DM*DM + 255)/256, blk, 0, stream>>>(qkv_w + (size_t)l*3*DM*DM, SQKV, (long long)3*DM*DM);
        cvt_f16<<<(DM*DM   + 255)/256, blk, 0, stream>>>(r_w   + (size_t)l*DM*DM,   SRW,  (long long)DM*DM);
        cvt_f16<<<(DM*DM   + 255)/256, blk, 0, stream>>>(o_w   + (size_t)l*DM*DM,   SOW,  (long long)DM*DM);
        cvt_f16<<<(DFF*DM  + 255)/256, blk, 0, stream>>>(ff_w1 + (size_t)l*DFF*DM,  SW1,  (long long)DFF*DM);
        cvt_f16<<<(DM*DFF  + 255)/256, blk, 0, stream>>>(ff_w2 + (size_t)l*DM*DFF,  SW2,  (long long)DM*DFF);

        // ---- attention ----
        // LN1 over concat([mems[l]; x])
        ln_kernel<<<TOKK, 128, 0, stream>>>(
            mems + (size_t)l*MLEN*BSZ*DM, X,
            ln1_g + (size_t)l*DM, ln1_b + (size_t)l*DM,
            H16, nullptr, MLEN, BSZ);
        // Fused QKV GEMM: (2560 x 3072 x 1024)
        gemm_wmma<0><<<dim3(3*DM/128, TOKK/64, 1), blk, 0, stream>>>(
            H16, SQKV, WH, nullptr, nullptr,
            TOKK, 3*DM, DM, DM, DM, 3*DM, 1, 0,0,0,0,0,0);
        // r_head_k = pos_emb @ r_w^T: (640 x 1024 x 1024)
        gemm_wmma<0><<<dim3(DM/128, KLEN/64, 1), blk, 0, stream>>>(
            POS, SRW, RK, nullptr, nullptr,
            KLEN, DM, DM, DM, DM, DM, 1, 0,0,0,0,0,0);
        // split Q/K/V, add relative biases, transpose V
        qkv_prep<<<((long long)TOKK*DM)/256, blk, 0, stream>>>(WH, rwb, rrb, QW, QR, KB, VT);

        // AC = (q+r_w_bias) @ k^T, 64-way batched (512 x 640 x 64)
        gemm_wmma<1><<<dim3(KLEN/128, QLEN/64, BSZ*NH), blk, 0, stream>>>(
            QW, KB, SC, nullptr, nullptr,
            QLEN, KLEN, DH, DH, DH, KLEN, NH,
            16*qz, qz, 16*kz, kz, 16*sz, sz);
        // BD = (q+r_r_bias) @ r_head_k^T with fused rel_shift scatter-add
        gemm_wmma<4><<<dim3(KLEN/128, QLEN/64, BSZ*NH), blk, 0, stream>>>(
            QR, RK, SC, nullptr, nullptr,
            QLEN, KLEN, DH, DH, DM, KLEN, NH,
            16*qz, qz, 0, DH, 16*sz, sz);
        // masked softmax (scale 1/sqrt(64)); prob (f16) written in place
        softmax_kernel<<<BSZ*NH*QLEN, 128, 0, stream>>>(SC, 0.125f);
        // attn_vec = prob @ v, batched (512 x 64 x 640); A = in-place f16 prob
        // N=64 < block tile 128 -> waves with tileN>=64 whole-wave exit.
        gemm_wmma<0><<<dim3(1, QLEN/64, BSZ*NH), blk, 0, stream>>>(
            (const _Float16*)SC, VT, AV, nullptr, nullptr,
            QLEN, DH, KLEN, 2*KLEN, KLEN, BSZ*DM, NH,
            32*sz, 2*sz, (long long)16*DH*KLEN, (long long)DH*KLEN, (long long)DM, (long long)DH);
        // x += attn_vec @ o_w^T (fused residual)
        gemm_wmma<3><<<dim3(DM/128, TOKQ/64, 1), blk, 0, stream>>>(
            AV, SOW, X, nullptr, X,
            TOKQ, DM, DM, DM, DM, DM, 1, 0,0,0,0,0,0);

        // ---- FFN ----
        ln_kernel<<<TOKQ, 128, 0, stream>>>(
            nullptr, X, ln2_g + (size_t)l*DM, ln2_b + (size_t)l*DM,
            H2, nullptr, 0, BSZ);
        // hidden = relu(h2 @ W1^T + b1): (2048 x 4096 x 1024)
        gemm_wmma<2><<<dim3(DFF/128, TOKQ/64, 1), blk, 0, stream>>>(
            H2, SW1, FFH, ff_b1 + (size_t)l*DFF, nullptr,
            TOKQ, DFF, DM, DM, DM, DFF, 1, 0,0,0,0,0,0);
        // x += hidden @ W2^T + b2: (2048 x 1024 x 4096)
        gemm_wmma<3><<<dim3(DM/128, TOKQ/64, 1), blk, 0, stream>>>(
            FFH, SW2, X, ff_b2 + (size_t)l*DM, X,
            TOKQ, DM, DFF, DFF, DFF, DM, 1, 0,0,0,0,0,0);
    }

    // final layernorm -> d_out (f32)
    ln_kernel<<<TOKQ, 128, 0, stream>>>(
        nullptr, X, fin_g, fin_b, nullptr, (float*)d_out, 0, BSZ);
}
